// LearnableKeyframeIndexSelector_90469191122984
// MI455X (gfx1250) — matrix operand order repaired and checked by reading, not compiled, not run on hardware
//
#include <hip/hip_runtime.h>
#include <hip/hip_bf16.h>

// Problem constants (from reference)
#define B_SZ   64
#define T_SZ   2048
#define D_SZ   512
#define HID_SZ 512
#define K_TOP  16
#define MROWS  (B_SZ * T_SZ)        // 131072 rows of motion (flattened)
#define ROWS_PER_BLK 32             // 2 M-tiles per block

typedef __attribute__((ext_vector_type(16))) __bf16 v16bf;
typedef __attribute__((ext_vector_type(8)))  float  v8f;

__device__ __forceinline__ unsigned short f2bf(float f) {
    // round-to-nearest-even f32 -> bf16
    unsigned int u = __float_as_uint(f);
    unsigned int r = u + 0x7FFFu + ((u >> 16) & 1u);
    return (unsigned short)(r >> 16);
}

// ---------------------------------------------------------------------------
// Kernel 1: W1 [K=512][N=512] f32  ->  W1t [N=512][K=512] bf16 (workspace)
// ---------------------------------------------------------------------------
__global__ void __launch_bounds__(256) kf_prep_w1t(const float* __restrict__ W1,
                                                   unsigned short* __restrict__ W1t) {
    int idx = blockIdx.x * 256 + threadIdx.x;   // 0 .. 262143
    int n = idx >> 9;
    int k = idx & 511;
    W1t[n * 512 + k] = f2bf(W1[k * 512 + n]);
}

// ---------------------------------------------------------------------------
// Kernel 2: fused  scores[m] = sum_n relu( (motion@W1)[m,n] + b1[n] ) * W2[n] + b2
// Block = 32 M-rows, 16 waves. Wave owns a 2 Mtile x 2 Ntile accumulator block:
//   - each B fragment feeds 2 WMMAs (halves W1t refetch traffic)
//   - B fragments are double-buffered one K-step ahead (hides L2 latency)
// ---------------------------------------------------------------------------
__global__ void __launch_bounds__(512) kf_score_gemm(
    const float*          __restrict__ motion,
    const unsigned short* __restrict__ W1t,     // bf16 [N][K]
    const float*          __restrict__ b1,
    const float*          __restrict__ W2,
    const float*          __restrict__ b2,
    float*                __restrict__ scores)
{
    __shared__ unsigned short sA[ROWS_PER_BLK * 520];  // 32 rows x 512 bf16, stride 520 (bank pad)
    __shared__ float sScore[ROWS_PER_BLK];

    const int tid  = threadIdx.x;
    const int row0 = blockIdx.x * ROWS_PER_BLK;

    // Stage 32x512 motion tile into LDS as bf16 (4096 float4 loads / block)
    const float* src = motion + (size_t)row0 * D_SZ;
    #pragma unroll
    for (int i = 0; i < 8; ++i) {
        int q  = tid + i * 512;                 // float4 slot 0..4095
        int r  = q >> 7;                        // row 0..31
        int c4 = q & 127;                       // float4 within row
        float4 f = ((const float4*)(src + (size_t)r * D_SZ))[c4];
        unsigned int p0 = (unsigned int)f2bf(f.x) | ((unsigned int)f2bf(f.y) << 16);
        unsigned int p1 = (unsigned int)f2bf(f.z) | ((unsigned int)f2bf(f.w) << 16);
        *(uint2*)&sA[r * 520 + c4 * 4] = make_uint2(p0, p1);
    }
    if (tid < ROWS_PER_BLK) sScore[tid] = 0.0f;
    __syncthreads();

    const int lane  = tid & 31;
    const int wave  = tid >> 5;                 // 0..15
    const int m     = lane & 15;                // tile row (A) / tile col (B,C)
    const int koffA = (lane >> 4) * 8;          // A: half-wave K sub-offset
    const int koffB = (lane >> 4) * 8;          // B: half-wave K sub-offset
    const int nBase = wave * 32;                // 2 N-tiles of 16 per wave

    float b1v[2], w2v[2];
    const unsigned short* bptr[2];
    #pragma unroll
    for (int nt = 0; nt < 2; ++nt) {
        int n = nBase + nt * 16 + m;            // absolute hidden-column for this lane
        b1v[nt]  = b1[n];
        w2v[nt]  = W2[n];
        bptr[nt] = W1t + (size_t)n * 512 + koffB;
    }

    v8f acc[2][2];                               // [mtile][ntile]
    #pragma unroll
    for (int mt = 0; mt < 2; ++mt)
        #pragma unroll
        for (int nt = 0; nt < 2; ++nt)
            #pragma unroll
            for (int j = 0; j < 8; ++j) acc[mt][nt][j] = 0.0f;

    const unsigned short* aBase[2];
    aBase[0] = &sA[(0 * 16 + m) * 520 + koffA];
    aBase[1] = &sA[(1 * 16 + m) * 520 + koffA];

    union Frag { v16bf v; uint4 q[2]; };
    Frag bcur[2], bnxt[2], afr[2];

    // Preload B fragments for k0 = 0
    #pragma unroll
    for (int nt = 0; nt < 2; ++nt) {
        bcur[nt].q[0] = *(const uint4*)(bptr[nt]);
        bcur[nt].q[1] = *(const uint4*)(bptr[nt] + 16);
    }

    #pragma unroll 4
    for (int k0 = 0; k0 < D_SZ; k0 += 32) {
        // A fragments for both M-tiles (LDS, low latency)
        #pragma unroll
        for (int mt = 0; mt < 2; ++mt) {
            afr[mt].q[0] = *(const uint4*)(aBase[mt] + k0);        // K = k0+koffA    ..+7
            afr[mt].q[1] = *(const uint4*)(aBase[mt] + k0 + 16);   // K = k0+koffA+16 ..+7
        }
        // Prefetch next K-step's B fragments (wrap-masked: branchless, in-bounds)
        const int k0n = (k0 + 32) & (D_SZ - 1);
        #pragma unroll
        for (int nt = 0; nt < 2; ++nt) {
            bnxt[nt].q[0] = *(const uint4*)(bptr[nt] + k0n);
            bnxt[nt].q[1] = *(const uint4*)(bptr[nt] + k0n + 16);
        }
        // 4 WMMAs consuming the previously-loaded B fragments
        #pragma unroll
        for (int nt = 0; nt < 2; ++nt)
            #pragma unroll
            for (int mt = 0; mt < 2; ++mt)
                acc[mt][nt] = __builtin_amdgcn_wmma_f32_16x16x32_bf16(
                    false, afr[mt].v, false, bcur[nt].v, (short)0, acc[mt][nt], false, false);
        bcur[0] = bnxt[0];
        bcur[1] = bnxt[1];
    }

    // Epilogue: relu(c + b1) * W2, reduce 16 columns (lanes) per row.
    float part[2][8];
    #pragma unroll
    for (int mt = 0; mt < 2; ++mt) {
        #pragma unroll
        for (int i = 0; i < 8; ++i) part[mt][i] = 0.0f;
        #pragma unroll
        for (int nt = 0; nt < 2; ++nt)
            #pragma unroll
            for (int i = 0; i < 8; ++i) {
                float c = acc[mt][nt][i] + b1v[nt];
                part[mt][i] += (c > 0.0f ? c : 0.0f) * w2v[nt];
            }
        #pragma unroll
        for (int i = 0; i < 8; ++i) {
            float v = part[mt][i];
            v += __shfl_xor(v, 1, 32);
            v += __shfl_xor(v, 2, 32);
            v += __shfl_xor(v, 4, 32);
            v += __shfl_xor(v, 8, 32);
            part[mt][i] = v;
        }
    }
    // C layout: VGPR i, lanes 0-15 -> row i, lanes 16-31 -> row 8+i
    if (m == 0) {
        int rbase = (lane >> 4) * 8;
        #pragma unroll
        for (int mt = 0; mt < 2; ++mt)
            #pragma unroll
            for (int i = 0; i < 8; ++i)
                atomicAdd(&sScore[mt * 16 + rbase + i], part[mt][i]);
    }
    __syncthreads();
    if (tid < ROWS_PER_BLK) scores[row0 + tid] = sScore[tid] + b2[0];
}

// ---------------------------------------------------------------------------
// Kernel 3: per-batch softmax over T, top-16 (tie -> lowest index), ST gather.
// one_hot_st row == ((1-p)+p) on the selected frame, exactly 0 elsewhere.
// ---------------------------------------------------------------------------
__global__ void __launch_bounds__(256) kf_softmax_topk_gather(
    const float* __restrict__ scores,
    const float* __restrict__ motion,
    float* __restrict__ out_kf,      // [B, K, D]
    float* __restrict__ out_idx,     // [B, K] (as float)
    float* __restrict__ out_probs)   // [B, T]
{
    __shared__ float sP[T_SZ];
    __shared__ float sRed[256];
    __shared__ int   sRedI[256];
    __shared__ float sScale[K_TOP];
    __shared__ int   sIdx[K_TOP];

    const int b   = blockIdx.x;
    const int tid = threadIdx.x;
    const float* sc = scores + (size_t)b * T_SZ;

    // ---- max ----
    float lv[8];
    float lmax = -3.4e38f;
    #pragma unroll
    for (int i = 0; i < 8; ++i) {
        float v = sc[tid + i * 256];
        lv[i] = v;
        lmax = fmaxf(lmax, v);
    }
    sRed[tid] = lmax;
    __syncthreads();
    for (int s = 128; s > 0; s >>= 1) {
        if (tid < s) sRed[tid] = fmaxf(sRed[tid], sRed[tid + s]);
        __syncthreads();
    }
    float gmax = sRed[0];
    __syncthreads();

    // ---- exp / sum ----
    float lsum = 0.0f;
    #pragma unroll
    for (int i = 0; i < 8; ++i) {
        float e = __expf(lv[i] - gmax);
        lv[i] = e;
        lsum += e;
    }
    sRed[tid] = lsum;
    __syncthreads();
    for (int s = 128; s > 0; s >>= 1) {
        if (tid < s) sRed[tid] += sRed[tid + s];
        __syncthreads();
    }
    float inv = 1.0f / sRed[0];
    __syncthreads();

    #pragma unroll
    for (int i = 0; i < 8; ++i) {
        int t = tid + i * 256;
        float p = lv[i] * inv;
        sP[t] = p;
        out_probs[(size_t)b * T_SZ + t] = p;
    }
    __syncthreads();

    // ---- iterative top-K argmax with masking ----
    for (int k = 0; k < K_TOP; ++k) {
        float bv = -1.0f; int bi = 0;
        #pragma unroll
        for (int i = 0; i < 8; ++i) {
            int t = tid + i * 256;
            float v = sP[t];
            if (v > bv || (v == bv && t < bi)) { bv = v; bi = t; }
        }
        sRed[tid] = bv; sRedI[tid] = bi;
        __syncthreads();
        for (int s = 128; s > 0; s >>= 1) {
            if (tid < s) {
                float v2 = sRed[tid + s]; int i2 = sRedI[tid + s];
                if (v2 > sRed[tid] || (v2 == sRed[tid] && i2 < sRedI[tid])) {
                    sRed[tid] = v2; sRedI[tid] = i2;
                }
            }
            __syncthreads();
        }
        if (tid == 0) {
            int idx = sRedI[0];
            float p = sP[idx];
            sIdx[k]   = idx;
            sScale[k] = (1.0f - p) + p;         // straight-through residual, fp32-exact
            sP[idx]   = -1.0f;                   // mask
            out_idx[b * K_TOP + k] = (float)idx;
        }
        __syncthreads();
    }

    // ---- gather key frames ----
    for (int q = tid; q < K_TOP * D_SZ; q += 256) {
        int k = q >> 9;
        int d = q & 511;
        out_kf[((size_t)b * K_TOP + k) * D_SZ + d] =
            sScale[k] * motion[((size_t)b * T_SZ + sIdx[k]) * D_SZ + d];
    }
}

// ---------------------------------------------------------------------------
extern "C" void kernel_launch(void* const* d_in, const int* in_sizes, int n_in,
                              void* d_out, int out_size, void* d_ws, size_t ws_size,
                              hipStream_t stream) {
    const float* motion = (const float*)d_in[0];   // [B,T,D]
    const float* W1     = (const float*)d_in[1];   // [D,HID]
    const float* b1     = (const float*)d_in[2];   // [HID]
    const float* W2     = (const float*)d_in[3];   // [HID,1]
    const float* b2     = (const float*)d_in[4];   // [1]

    float* out = (float*)d_out;
    float* out_kf    = out;                                   // B*K*D
    float* out_idx   = out + (size_t)B_SZ * K_TOP * D_SZ;     // B*K
    float* out_probs = out_idx + (size_t)B_SZ * K_TOP;        // B*T

    // Workspace: scores f32 (512 KB) | W1t bf16 (512 KB)
    float*          scores = (float*)d_ws;
    unsigned short* W1t    = (unsigned short*)((char*)d_ws + (size_t)MROWS * sizeof(float));

    kf_prep_w1t<<<(D_SZ * HID_SZ) / 256, 256, 0, stream>>>(W1, W1t);
    kf_score_gemm<<<MROWS / ROWS_PER_BLK, 512, 0, stream>>>(motion, W1t, b1, W2, b2, scores);
    kf_softmax_topk_gather<<<B_SZ, 256, 0, stream>>>(scores, motion, out_kf, out_idx, out_probs);
}
